// HardActor_44461501448356
// MI455X (gfx1250) — matrix operand order repaired and checked by reading, not compiled, not run on hardware
//
#include <hip/hip_runtime.h>
#include <hip/hip_bf16.h>
#include <math.h>

typedef _Float16 v16h __attribute__((ext_vector_type(16)));
typedef _Float16 v8h  __attribute__((ext_vector_type(8)));
typedef _Float16 v4h  __attribute__((ext_vector_type(4)));
typedef float    v8f  __attribute__((ext_vector_type(8)));

#define B_TOTAL   65536
#define DIM       256     // input features == HIDDEN
#define HID       256
#define A_DIM     128
#define NREG      4

#define ROWS_PER_BLOCK 64
#define WAVES_PER_BLOCK 4
#define THREADS (WAVES_PER_BLOCK * 32)

// ---------------------------------------------------------------------------
// Prep: out[n*K + k] = (f16) in[k*N + n]   (transpose + fp32->fp16 convert)
// ---------------------------------------------------------------------------
__global__ void transpose_f32_to_f16(const float* __restrict__ in,
                                     _Float16* __restrict__ out,
                                     int K, int N) {
  int idx = blockIdx.x * blockDim.x + threadIdx.x;
  if (idx >= K * N) return;
  int n = idx / K;
  int k = idx - n * K;
  out[idx] = (_Float16)in[(size_t)k * N + n];
}

// ---------------------------------------------------------------------------
// Fragment loaders (per the CDNA5 ISA 16-bit operand layouts)
// ---------------------------------------------------------------------------
__device__ __forceinline__ v16h load_frag_A(const _Float16* base) {
  // halfs [0..7] at +0, halfs [8..15] at +16 (K offsets kb..kb+7, kb+16..kb+23)
  v8h lo = *(const v8h*)(base);
  v8h hi = *(const v8h*)(base + 16);
  return __builtin_shufflevector(lo, hi, 0,1,2,3,4,5,6,7,8,9,10,11,12,13,14,15);
}

__device__ __forceinline__ v16h load_frag_B(const _Float16* base) {
  // halfs [0..7] at +0, halfs [8..15] at +16 (K offsets kb..kb+7, kb+16..kb+23)
  v8h lo = *(const v8h*)(base);
  v8h hi = *(const v8h*)(base + 16);
  return __builtin_shufflevector(lo, hi, 0,1,2,3,4,5,6,7,8,9,10,11,12,13,14,15);
}

// ---------------------------------------------------------------------------
// Fused actor kernel: x -> relu(xW1+b1) -> relu(hW2+b2) -> regime head -> out
// ---------------------------------------------------------------------------
__global__ __launch_bounds__(THREADS)
void actor_fused(const float* __restrict__ x,
                 const float* __restrict__ b1,
                 const float* __restrict__ b2,
                 const float* __restrict__ bh,
                 const float* __restrict__ log_std,
                 const _Float16* __restrict__ W1t,   // [256][256] N-major
                 const _Float16* __restrict__ W2t,   // [256][256] N-major
                 const _Float16* __restrict__ Wht,   // [4][128][256] N-major
                 float* __restrict__ out_mean,
                 float* __restrict__ out_std) {
  __shared__ __align__(16) _Float16 bufA[ROWS_PER_BLOCK * HID]; // 32 KB
  __shared__ __align__(16) _Float16 bufB[ROWS_PER_BLOCK * HID]; // 32 KB

  const int t    = threadIdx.x;
  const int wave = t >> 5;
  const int lane = t & 31;
  const int nl   = lane & 15;      // N index within 16-wide tile
  const int hs   = lane >> 4;      // half-select (0 or 1)
  const int blockRow = blockIdx.x * ROWS_PER_BLOCK;

  // ---- stage x tile [64 x 256] fp32 -> fp16 into bufA (coalesced) ----
  {
    const int vecPerRow = DIM / 4;                       // 64 float4 per row
    const int total = ROWS_PER_BLOCK * vecPerRow;        // 4096
    for (int i = t; i < total; i += THREADS) {
      int row = i / vecPerRow;
      int c4  = i - row * vecPerRow;
      float4 v = ((const float4*)(x + (size_t)(blockRow + row) * DIM))[c4];
      v4h h;
      h[0] = (_Float16)v.x; h[1] = (_Float16)v.y;
      h[2] = (_Float16)v.z; h[3] = (_Float16)v.w;
      *(v4h*)&bufA[row * DIM + c4 * 4] = h;
    }
  }
  __syncthreads();
  // From here on each wave only touches its own 16 rows of bufA/bufB.

  const int r0   = wave * 16;            // local row base of this wave
  const int aRow = r0 + nl;              // A-operand row for this lane
  const int kbA  = hs ? 8  : 0;          // A-fragment K base
  const int kbB  = hs ? 8  : 0;          // B-fragment K base

  // Regimes for the 8 rows this lane's C/D fragment covers (x unchanged in HBM)
  int reg[8];
#pragma unroll
  for (int v = 0; v < 8; ++v) {
    int row = blockRow + r0 + v + 8 * hs;
    reg[v] = (int)x[(size_t)row * DIM + (DIM - 1)];
  }

  // ---- GEMM1: h = relu(x @ W1 + b1), bufA -> bufB ----
  for (int n = 0; n < HID / 16; ++n) {
    float bias = b1[n * 16 + nl];
    v8f c;
#pragma unroll
    for (int v = 0; v < 8; ++v) c[v] = bias;
    const _Float16* wbase = W1t + (size_t)(n * 16 + nl) * DIM + kbB;
    const _Float16* abase = &bufA[aRow * DIM + kbA];
#pragma unroll
    for (int kk = 0; kk < DIM / 32; ++kk) {
      v16h a = load_frag_A(abase + kk * 32);
      v16h b = load_frag_B(wbase + kk * 32);
      c = __builtin_amdgcn_wmma_f32_16x16x32_f16(false, a, false, b,
                                                 (short)0, c, false, false);
    }
#pragma unroll
    for (int v = 0; v < 8; ++v) {
      float r = c[v] > 0.f ? c[v] : 0.f;
      bufB[(r0 + v + 8 * hs) * HID + n * 16 + nl] = (_Float16)r;
    }
  }

  // ---- GEMM2: feats = relu(h @ W2 + b2), bufB -> bufA ----
  for (int n = 0; n < HID / 16; ++n) {
    float bias = b2[n * 16 + nl];
    v8f c;
#pragma unroll
    for (int v = 0; v < 8; ++v) c[v] = bias;
    const _Float16* wbase = W2t + (size_t)(n * 16 + nl) * HID + kbB;
    const _Float16* abase = &bufB[aRow * HID + kbA];
#pragma unroll
    for (int kk = 0; kk < HID / 32; ++kk) {
      v16h a = load_frag_A(abase + kk * 32);
      v16h b = load_frag_B(wbase + kk * 32);
      c = __builtin_amdgcn_wmma_f32_16x16x32_f16(false, a, false, b,
                                                 (short)0, c, false, false);
    }
#pragma unroll
    for (int v = 0; v < 8; ++v) {
      float r = c[v] > 0.f ? c[v] : 0.f;
      bufA[(r0 + v + 8 * hs) * HID + n * 16 + nl] = (_Float16)r;
    }
  }

  // ---- Regime heads: mean = 0.1 * (feats @ Wh[g] + bh[g]), hard-routed ----
  for (int g = 0; g < NREG; ++g) {
    int want = 0;
#pragma unroll
    for (int v = 0; v < 8; ++v) want |= (reg[v] == g) ? 1 : 0;
    if (__ballot(want) == 0ull) continue;   // regime absent in this wave's rows

    const _Float16* wg = Wht + (size_t)g * A_DIM * HID;
    for (int n = 0; n < A_DIM / 16; ++n) {
      float bias = bh[g * A_DIM + n * 16 + nl];
      v8f c;
#pragma unroll
      for (int v = 0; v < 8; ++v) c[v] = bias;
      const _Float16* wbase = wg + (size_t)(n * 16 + nl) * HID + kbB;
      const _Float16* abase = &bufA[aRow * HID + kbA];
#pragma unroll
      for (int kk = 0; kk < HID / 32; ++kk) {
        v16h a = load_frag_A(abase + kk * 32);
        v16h b = load_frag_B(wbase + kk * 32);
        c = __builtin_amdgcn_wmma_f32_16x16x32_f16(false, a, false, b,
                                                   (short)0, c, false, false);
      }
#pragma unroll
      for (int v = 0; v < 8; ++v) {
        if (reg[v] == g) {
          int row = blockRow + r0 + v + 8 * hs;
          out_mean[(size_t)row * A_DIM + n * 16 + nl] = 0.1f * c[v];
        }
      }
    }
  }

  // ---- std = clip(exp(log_std), 1e-3, 1), broadcast (coalesced) ----
  {
    const int total = ROWS_PER_BLOCK * A_DIM; // 8192
    for (int i = t; i < total; i += THREADS) {
      int row = i >> 7;
      int col = i & (A_DIM - 1);
      float sv = expf(log_std[col]);
      sv = fminf(fmaxf(sv, 1e-3f), 1.0f);
      out_std[(size_t)(blockRow + row) * A_DIM + col] = sv;
    }
  }
}

// ---------------------------------------------------------------------------
extern "C" void kernel_launch(void* const* d_in, const int* in_sizes, int n_in,
                              void* d_out, int out_size, void* d_ws, size_t ws_size,
                              hipStream_t stream) {
  (void)in_sizes; (void)n_in; (void)out_size; (void)ws_size;

  const float* x  = (const float*)d_in[0];
  const float* W1 = (const float*)d_in[1];
  const float* b1 = (const float*)d_in[2];
  const float* W2 = (const float*)d_in[3];
  const float* b2 = (const float*)d_in[4];
  const float* Wh = (const float*)d_in[5];
  const float* bh = (const float*)d_in[6];
  const float* ls = (const float*)d_in[7];

  _Float16* W1t = (_Float16*)d_ws;                 // [256][256] f16 (128 KB)
  _Float16* W2t = W1t + (size_t)DIM * HID;         // [256][256] f16 (128 KB)
  _Float16* Wht = W2t + (size_t)HID * HID;         // [4][128][256] f16 (256 KB)

  float* out_mean = (float*)d_out;
  float* out_std  = out_mean + (size_t)B_TOTAL * A_DIM;

  // Weight prep: transpose + f32->f16 (tiny; weights then L2-resident)
  {
    int n1 = DIM * HID;
    transpose_f32_to_f16<<<(n1 + 255) / 256, 256, 0, stream>>>(W1, W1t, DIM, HID);
    transpose_f32_to_f16<<<(n1 + 255) / 256, 256, 0, stream>>>(W2, W2t, HID, HID);
    int nh = HID * A_DIM;
    for (int g = 0; g < NREG; ++g) {
      transpose_f32_to_f16<<<(nh + 255) / 256, 256, 0, stream>>>(
          Wh + (size_t)g * HID * A_DIM, Wht + (size_t)g * A_DIM * HID, HID, A_DIM);
    }
  }

  actor_fused<<<B_TOTAL / ROWS_PER_BLOCK, THREADS, 0, stream>>>(
      x, b1, b2, bh, ls, W1t, W2t, Wht, out_mean, out_std);
}
